// DynamicTreeDraftingLoopWrapper_12197707121152
// MI455X (gfx1250) — matrix operand order repaired and checked by reading, not compiled, not run on hardware
//
#include <hip/hip_runtime.h>
#include <hip/hip_bf16.h>

// ---------------------------------------------------------------------------
// Dynamic tree drafting loop for MI455X (gfx1250).
//   B=8, V=32000, D=1024, K=10, L=4, T=63, HIST=310.
// Strategy: convert w_lm to bf16 WMMA-B packed layout once per call (65.5MB,
// L2-resident afterwards); per layer gather the <=80 embedding rows into the
// WMMA-A packed layout, run a bf16 WMMA GEMM with the whole A panel in LDS
// (staged via GLOBAL_LOAD_ASYNC_TO_LDS when available), then log-softmax +
// hierarchical top-k selection kernels.
// ---------------------------------------------------------------------------

#define V_SZ   32000
#define D_SZ   1024
#define B_SZ   8
#define K_TOP  10
#define T_DRAFT 63
#define HIST   310          // K + 3*K*K
#define PARW   31           // 1 + (L-1)*K
#define KC     32           // D / 32 (k-chunks of 32 for 16x16x32 wmma)
#define NT     2000         // V / 16 (n-tiles)

typedef __attribute__((ext_vector_type(16))) __bf16 v16bf;
typedef __attribute__((ext_vector_type(8)))  float  v8f;

#if defined(__HIP_DEVICE_COMPILE__) && \
    __has_builtin(__builtin_amdgcn_global_load_async_to_lds_b128) && \
    __has_builtin(__builtin_amdgcn_s_wait_asynccnt)
#define HAVE_ASYNC_LDS 1
#else
#define HAVE_ASYNC_LDS 0
#endif

__device__ __forceinline__ __bf16 f2bf(float f) {
    unsigned int u = __float_as_uint(f);
    u += 0x7FFFu + ((u >> 16) & 1u);            // round-to-nearest-even
    unsigned short h = (unsigned short)(u >> 16);
    return __builtin_bit_cast(__bf16, h);
}

__device__ __forceinline__ float neg_inf() { return -__builtin_inff(); }

// ---------------------------------------------------------------------------
// Pack w_lm [D,V] f32 row-major -> bf16 WMMA B-operand layout:
// block index = n_tile*32 + kc ; inside: 32 lanes x 8 dwords (16 bf16/lane).
// lane l: col n = n_tile*16 + (l&15), half = l>>4 ;
// dword j holds K = half*16 + 2j, 2j+1 of the 32-wide k-chunk.
// ---------------------------------------------------------------------------
__global__ __launch_bounds__(256)
void pack_w_kernel(const float* __restrict__ w, __bf16* __restrict__ Wp) {
    const int bid    = blockIdx.x;        // 0 .. NT*KC-1
    const int n_tile = bid >> 5;
    const int kc     = bid & 31;
    const int t      = threadIdx.x;
    const int n      = t & 15;
    const int kk     = t >> 4;            // 0..15 (k-pair index)
    const int j      = kk & 7;
    const int half   = kk >> 3;
    const int lane   = half * 16 + n;
    const int klocal = half * 16 + 2 * j;
    const int krow   = kc * 32 + klocal;
    const int col    = n_tile * 16 + n;
    const float w0 = w[(size_t)krow * V_SZ + col];
    const float w1 = w[(size_t)(krow + 1) * V_SZ + col];
    const size_t o = ((((size_t)n_tile * KC + kc) * 32 + lane) * 16) + 2 * j;
    Wp[o]     = f2bf(w0);
    Wp[o + 1] = f2bf(w1);
}

// ---------------------------------------------------------------------------
// Gather embedding rows for the current frontier tokens into the bf16 WMMA
// A-operand layout (16-bit A 16x32, contiguous k order, same as packed B):
// lane l: m = l&15, half = l>>4 ; dword j holds
//   K = half*16 + 2*j (+1).
// Rows m >= M are zero-padded.
// ---------------------------------------------------------------------------
__global__ __launch_bounds__(256)
void gather_a_kernel(const int* __restrict__ tokens, int M,
                     const float* __restrict__ embed,
                     __bf16* __restrict__ Ap) {
    const int bid    = blockIdx.x;        // m_tile*KC + kc
    const int m_tile = bid >> 5;
    const int kc     = bid & 31;
    const int t      = threadIdx.x;
    const int lane   = t & 31;
    const int j      = t >> 5;            // 0..7
    const int half   = lane >> 4;
    const int m_loc  = lane & 15;
    const int klocal = half * 16 + 2 * j;
    const int m      = m_tile * 16 + m_loc;
    float e0 = 0.0f, e1 = 0.0f;
    if (m < M) {
        const int tok = tokens[m];
        const size_t base = (size_t)tok * D_SZ + kc * 32 + klocal;
        e0 = embed[base];
        e1 = embed[base + 1];
    }
    const size_t o = ((((size_t)m_tile * KC + kc) * 32 + lane) * 16) + 2 * j;
    Ap[o]     = f2bf(e0);
    Ap[o + 1] = f2bf(e1);
}

// ---------------------------------------------------------------------------
// GEMM: logits[MT*16, V] = A(bf16) @ W(bf16), f32 accumulate via
// v_wmma_f32_16x16x32_bf16. 256 threads = 8 waves; each wave owns one n-tile,
// block covers 8 consecutive n-tiles; whole packed A panel staged in LDS
// (async direct-to-LDS when available) and reused across all MT m-tiles per
// B k-chunk load (one 32B contiguous load per lane).
// ---------------------------------------------------------------------------
template <int MT>
__global__ __launch_bounds__(256)
void gemm_wmma_kernel(const __bf16* __restrict__ Wp,
                      const __bf16* __restrict__ Ap,
                      float* __restrict__ logits) {
    extern __shared__ __bf16 sA[];                 // MT * KC * 32 * 16 elems
    const int tid    = threadIdx.x;
    const int lane   = tid & 31;
    const int wave   = tid >> 5;                   // 0..7
    const int n_tile = blockIdx.x * 8 + wave;      // 0..NT-1

    // Stage packed A panel into LDS (MT * 32 KB).
#if HAVE_ASYNC_LDS
    {
        typedef int tdm4i __attribute__((__vector_size__(16)));
        typedef __attribute__((address_space(1))) tdm4i* gptr_t;
        typedef __attribute__((address_space(3))) tdm4i* lptr_t;
        gptr_t g = (gptr_t)(void*)(__bf16*)Ap;     // AS(1) 16B-vector pointer
        lptr_t l = (lptr_t)(void*)sA;              // AS(3) 16B-vector pointer
        const int total = MT * 2048;               // 16-byte chunks
        for (int i = tid; i < total; i += 256)
            __builtin_amdgcn_global_load_async_to_lds_b128(g + i, l + i, 0, 0);
        __builtin_amdgcn_s_wait_asynccnt(0);
    }
#else
    {
        const uint4* src = (const uint4*)Ap;
        uint4*       dst = (uint4*)sA;
        const int total = MT * 2048;               // uint4 count
        for (int i = tid; i < total; i += 256) dst[i] = src[i];
    }
#endif
    __syncthreads();

    v8f acc[MT];
#pragma unroll
    for (int m = 0; m < MT; ++m)
#pragma unroll
        for (int j = 0; j < 8; ++j) acc[m][j] = 0.0f;

    const __bf16* wrow = Wp + ((size_t)n_tile * (KC * 32) + lane) * 16;

#pragma unroll 4
    for (int kc = 0; kc < KC; ++kc) {
        const v16bf b = *(const v16bf*)(wrow + (size_t)kc * 512);
        if (kc + 2 < KC)   // pull next chunks into the near (WGP) cache
            __builtin_prefetch((const void*)(wrow + (size_t)(kc + 2) * 512), 0, 3);
#pragma unroll
        for (int m = 0; m < MT; ++m) {
            const v16bf a = *(const v16bf*)(&sA[(((size_t)m * KC + kc) * 32 + lane) * 16]);
            acc[m] = __builtin_amdgcn_wmma_f32_16x16x32_bf16(
                false, a, false, b, (short)0, acc[m], false, false);
        }
    }

    // C/D 32-bit 16x16 layout: VGPR j, lane l -> row m = j + 8*(l>>4), col = l&15.
    const int col = n_tile * 16 + (lane & 15);
    const int mo  = 8 * (lane >> 4);
#pragma unroll
    for (int m = 0; m < MT; ++m)
#pragma unroll
        for (int j = 0; j < 8; ++j)
            logits[(size_t)(m * 16 + mo + j) * V_SZ + col] = acc[m][j];
}

// ---------------------------------------------------------------------------
// Per-row logsumexp over V (for log-softmax shift). One 256-thread block/row.
// ---------------------------------------------------------------------------
__global__ __launch_bounds__(256)
void lse_kernel(const float* __restrict__ logits, float* __restrict__ logZ) {
    const int row = blockIdx.x;
    const float* x = logits + (size_t)row * V_SZ;
    __shared__ float red[256];
    const int t = threadIdx.x;
    float m = neg_inf();
    for (int i = t; i < V_SZ; i += 256) m = fmaxf(m, x[i]);
    red[t] = m; __syncthreads();
    for (int s = 128; s > 0; s >>= 1) {
        if (t < s) red[t] = fmaxf(red[t], red[t + s]);
        __syncthreads();
    }
    const float gm = red[0]; __syncthreads();
    float sum = 0.0f;
    for (int i = t; i < V_SZ; i += 256) sum += expf(x[i] - gm);
    red[t] = sum; __syncthreads();
    for (int s = 128; s > 0; s >>= 1) {
        if (t < s) red[t] += red[t + s];
        __syncthreads();
    }
    if (t == 0) logZ[row] = gm + logf(red[0]);
}

// ---------------------------------------------------------------------------
// Top-10 per row via 10 masked argmax passes (logits are scratch: winners are
// overwritten with -inf). Writes tokens and cumulative scores straight into
// the history buffers: score = logit - logZ + bias(parent cumulative score).
// ---------------------------------------------------------------------------
__global__ __launch_bounds__(256)
void topk10_kernel(float* __restrict__ logits, const float* __restrict__ logZ,
                   const float* __restrict__ bias, int rows_per_b, int base,
                   int* __restrict__ hist_tok, float* __restrict__ hist_sc) {
    const int row = blockIdx.x;
    float* x = logits + (size_t)row * V_SZ;
    __shared__ float sv[256];
    __shared__ int   si[256];
    const int t = threadIdx.x;
    const float lz = logZ[row];
    const float bv = bias ? bias[row] : 0.0f;
    const int b  = row / rows_per_b;
    const int jj = row % rows_per_b;
    const int obase = b * HIST + base + jj * K_TOP;

    for (int pass = 0; pass < K_TOP; ++pass) {
        float best = neg_inf(); int bidx = 0x7fffffff;
        for (int i = t; i < V_SZ; i += 256) {
            const float v = x[i];
            if (v > best || (v == best && i < bidx)) { best = v; bidx = i; }
        }
        sv[t] = best; si[t] = bidx; __syncthreads();
        for (int s = 128; s > 0; s >>= 1) {
            if (t < s) {
                if (sv[t + s] > sv[t] || (sv[t + s] == sv[t] && si[t + s] < si[t])) {
                    sv[t] = sv[t + s]; si[t] = si[t + s];
                }
            }
            __syncthreads();
        }
        if (t == 0) {
            const int w = si[0];
            hist_tok[obase + pass] = w;
            hist_sc[obase + pass]  = sv[0] - lz + bv;
            x[w] = neg_inf();                      // mask for next pass
        }
        __threadfence();
        __syncthreads();
    }
}

// ---------------------------------------------------------------------------
// Layer-0 frontier init: cur = top-10 roots, sel_pos = 0..9, parents[:,0]=-1.
// ---------------------------------------------------------------------------
__global__ void select0_kernel(const int* __restrict__ ht, const float* __restrict__ hs,
                               int* cur_tok, float* cur_sc, int* sel_pos, int* parents) {
    const int t = threadIdx.x;
    if (t < B_SZ * K_TOP) {
        const int b = t / K_TOP, i = t % K_TOP;
        cur_tok[t] = ht[b * HIST + i];
        cur_sc[t]  = hs[b * HIST + i];
        sel_pos[t] = i;
    }
    if (t < B_SZ) parents[t * PARW] = -1;
}

// ---------------------------------------------------------------------------
// Frontier re-selection for layers >= 1: top-10 of the 100 cumulative child
// scores per batch; track parent history positions. Tiny -> thread 0 serial.
// ---------------------------------------------------------------------------
__global__ void select_kernel(const int* __restrict__ ht, const float* __restrict__ hs,
                              int layer, int* cur_tok, float* cur_sc,
                              int* sel_pos, int* parents) {
    const int b = blockIdx.x;
    if (threadIdx.x != 0) return;
    const int base = K_TOP + (layer - 1) * K_TOP * K_TOP;
    float comb[K_TOP * K_TOP];
    for (int i = 0; i < K_TOP * K_TOP; ++i) comb[i] = hs[b * HIST + base + i];
    int nt[K_TOP]; float ns[K_TOP]; int np[K_TOP]; int nsp[K_TOP];
    for (int i = 0; i < K_TOP; ++i) {
        int bi = 0; float bvv = comb[0];
        for (int c = 1; c < K_TOP * K_TOP; ++c)
            if (comb[c] > bvv) { bvv = comb[c]; bi = c; }
        ns[i]  = bvv;
        nt[i]  = ht[b * HIST + base + bi];
        np[i]  = sel_pos[b * K_TOP + bi / K_TOP];   // parent history position
        nsp[i] = base + bi;                         // new frontier position
        comb[bi] = neg_inf();
    }
    for (int i = 0; i < K_TOP; ++i) {
        cur_tok[b * K_TOP + i] = nt[i];
        cur_sc[b * K_TOP + i]  = ns[i];
        parents[b * PARW + 1 + (layer - 1) * K_TOP + i] = np[i];
    }
    for (int i = 0; i < K_TOP; ++i) sel_pos[b * K_TOP + i] = nsp[i];
}

// ---------------------------------------------------------------------------
// Final: per batch top-63 of 310 history scores; emit all four outputs
// (as float, concatenated flat in reference return order).
// ---------------------------------------------------------------------------
__global__ __launch_bounds__(64)
void final_kernel(const int* __restrict__ ht, const float* __restrict__ hs,
                  const int* __restrict__ parents, float* __restrict__ out) {
    const int b = blockIdx.x;
    const int t = threadIdx.x;
    __shared__ float s[HIST];
    __shared__ float rv[64];
    __shared__ int   ri[64];
    for (int i = t; i < HIST; i += 64) s[i] = hs[b * HIST + i];
    __syncthreads();
    for (int k = 0; k < T_DRAFT; ++k) {
        float best = neg_inf(); int bi = 0x7fffffff;
        for (int i = t; i < HIST; i += 64) {
            const float v = s[i];
            if (v > best || (v == best && i < bi)) { best = v; bi = i; }
        }
        rv[t] = best; ri[t] = bi; __syncthreads();
        for (int st = 32; st > 0; st >>= 1) {
            if (t < st) {
                if (rv[t + st] > rv[t] || (rv[t + st] == rv[t] && ri[t + st] < ri[t])) {
                    rv[t] = rv[t + st]; ri[t] = ri[t + st];
                }
            }
            __syncthreads();
        }
        if (t == 0) {
            const int w = ri[0];
            out[(size_t)k * B_SZ + b]              = (float)ht[b * HIST + w]; // new_draft_tokens [T,B]
            out[T_DRAFT * B_SZ + b * T_DRAFT + k]  = rv[0];                   // final_sc [B,T]
            out[2 * T_DRAFT * B_SZ + b * T_DRAFT + k] = (float)w;             // indices [B,T]
            s[w] = neg_inf();
        }
        __syncthreads();
    }
    if (t < PARW)
        out[3 * T_DRAFT * B_SZ + b * PARW + t] = (float)parents[b * PARW + t]; // parent_buffer [B,31]
}

// ---------------------------------------------------------------------------
extern "C" void kernel_launch(void* const* d_in, const int* in_sizes, int n_in,
                              void* d_out, int out_size, void* d_ws, size_t ws_size,
                              hipStream_t stream) {
    const int*   input_ids = (const int*)d_in[0];    // [8]
    const float* embed     = (const float*)d_in[1];  // [32000,1024]
    const float* w_lm      = (const float*)d_in[2];  // [1024,32000]
    float* out = (float*)d_out;

    char* ws = (char*)d_ws;
    size_t off = 0;
    auto take = [&](size_t bytes) -> char* {
        char* p = ws + off;
        off += (bytes + 255) & ~(size_t)255;
        return p;
    };
    __bf16* Wp       = (__bf16*)take((size_t)NT * KC * 32 * 16 * 2);   // 65.5 MB packed B
    __bf16* Ap       = (__bf16*)take((size_t)5 * KC * 32 * 16 * 2);    // 160 KB packed A
    float*  logits   = (float*)take((size_t)80 * V_SZ * 4);            // 10.2 MB
    float*  logZ     = (float*)take(80 * 4);
    int*    hist_tok = (int*)take(B_SZ * HIST * 4);
    float*  hist_sc  = (float*)take(B_SZ * HIST * 4);
    int*    cur_tok  = (int*)take(80 * 4);
    float*  cur_sc   = (float*)take(80 * 4);
    int*    sel_pos  = (int*)take(80 * 4);
    int*    parents  = (int*)take(B_SZ * PARW * 4);
    (void)in_sizes; (void)n_in; (void)out_size; (void)ws_size;

    // One-time (per call) weight repack: f32 -> bf16 WMMA B layout.
    pack_w_kernel<<<NT * KC, 256, 0, stream>>>(w_lm, Wp);

    // ---- Layer 0: 8 root rows (1 m-tile, zero-padded to 16) ----
    gather_a_kernel<<<1 * KC, 256, 0, stream>>>(input_ids, B_SZ, embed, Ap);
    gemm_wmma_kernel<1><<<NT / 8, 256, 1 * KC * 32 * 16 * 2, stream>>>(Wp, Ap, logits);
    lse_kernel<<<B_SZ, 256, 0, stream>>>(logits, logZ);
    topk10_kernel<<<B_SZ, 256, 0, stream>>>(logits, logZ, nullptr, 1, 0,
                                            hist_tok, hist_sc);
    select0_kernel<<<1, 128, 0, stream>>>(hist_tok, hist_sc,
                                          cur_tok, cur_sc, sel_pos, parents);

    // ---- Layers 1..3: 80 frontier rows (5 m-tiles) ----
    for (int layer = 1; layer < 4; ++layer) {
        gather_a_kernel<<<5 * KC, 256, 0, stream>>>(cur_tok, B_SZ * K_TOP, embed, Ap);
        gemm_wmma_kernel<5><<<NT / 8, 256, 5 * KC * 32 * 16 * 2, stream>>>(Wp, Ap, logits);
        lse_kernel<<<B_SZ * K_TOP, 256, 0, stream>>>(logits, logZ);
        topk10_kernel<<<B_SZ * K_TOP, 256, 0, stream>>>(
            logits, logZ, cur_sc, K_TOP, K_TOP + (layer - 1) * K_TOP * K_TOP,
            hist_tok, hist_sc);
        select_kernel<<<B_SZ, 32, 0, stream>>>(hist_tok, hist_sc, layer,
                                               cur_tok, cur_sc, sel_pos, parents);
    }

    // ---- Final resample: global top-63 over history ----
    final_kernel<<<B_SZ, 64, 0, stream>>>(hist_tok, hist_sc, parents, out);
}
